// BaseAttention_46548855554192
// MI455X (gfx1250) — compile-verified, matrix operand order checked by reading, not run on hardware
//
#include <hip/hip_runtime.h>
#include <cstdint>

// ---------------------------------------------------------------------------
// Problem constants (reference: B=2, S=2048, H=2048, NH=16, HD=128)
// ---------------------------------------------------------------------------
#define BB 2
#define SS 2048
#define HH 2048
#define NHH 16
#define HDD 128
#define H3 (3 * HH)
#define MM (BB * SS) // 4096 rows of the token matrix

typedef __attribute__((ext_vector_type(16))) __bf16 v16bf;
typedef __attribute__((ext_vector_type(8))) float v8f;

union FragB {
  v16bf v;
  uint32_t u[8];
};
union FragF {
  v8f v;
  float f[8];
};

__device__ __forceinline__ uint16_t f32_to_bf16(float x) {
  uint32_t u = __float_as_uint(x);
  u = (u + 0x7FFFu + ((u >> 16) & 1u)) >> 16; // round-to-nearest-even
  return (uint16_t)u;
}

__device__ __forceinline__ v8f wmma_bf16(const v16bf& a, const v16bf& b,
                                         const v8f& c) {
  // (neg_a, A, neg_b, B, c_mod, C, reuse_a, reuse_b)
  return __builtin_amdgcn_wmma_f32_16x16x32_bf16(false, a, false, b, (short)0,
                                                 c, false, false);
}

// ---------------------------------------------------------------------------
// CDNA5 async global->LDS copy (ISA 15.18.3 op 98, tracked by ASYNCcnt).
// VDST VGPR holds the LDS byte address == low 32 bits of a generic LDS ptr.
// ---------------------------------------------------------------------------
__device__ __forceinline__ void async_ld_b128(void* lds, const void* g) {
  const uint32_t l = (uint32_t)(uintptr_t)lds;
  asm volatile("global_load_async_to_lds_b128 %0, %1, off"
               :
               : "v"(l), "v"(g)
               : "memory");
}
__device__ __forceinline__ void async_wait0() {
  asm volatile("s_wait_asynccnt 0x0" ::: "memory");
}

// Load one 16x32 bf16 A/B fragment from an LDS tile stored row-major
// [rows][ldk] with K contiguous. Per ISA 16-bit layout:
//   M/N = lane%16 ; K = (g>=4?16:0) + (lane>=16?8:0) + 2*(g&3) (+pair)
__device__ __forceinline__ void ld_frag(FragB& f, const uint16_t* lds, int row0,
                                        int ldk, int k0, int lane) {
  const int r = row0 + (lane & 15);
  const int kadd = (lane & 16) ? 8 : 0;
  const uint16_t* base = lds + r * ldk + k0;
#pragma unroll
  for (int g = 0; g < 8; ++g) {
    const int k = ((g & 4) << 2) + kadd + ((g & 3) << 1);
    f.u[g] = *(const uint32_t*)(base + k);
  }
}

// ---------------------------------------------------------------------------
// Kernel 1/4: GEMM  C[M,N] = A[M,K] @ Bw[N,K]^T (+bias)
// Block = 256 threads (8 waves). Block tile 128x128, wave tile 32x64.
// A is f32 (converted on the fly) or bf16; Bw is f32 [N][K] row-major,
// which is already the K-contiguous [N][K] layout the B-fragment wants.
// ---------------------------------------------------------------------------
template <bool A_BF16, bool ADD_BIAS>
__global__ __launch_bounds__(256) void gemm_wmma_kernel(
    const void* __restrict__ Aptr, const float* __restrict__ Bw,
    const float* __restrict__ bias, float* __restrict__ Cout, int M, int N,
    int K) {
  __shared__ __align__(16) uint16_t lA[128 * 32];
  __shared__ __align__(16) uint16_t lB[128 * 32];
  const int tid = threadIdx.x;
  const int lane = tid & 31;
  const int wave = tid >> 5;
  const int wm = wave >> 1; // 0..3 -> 32-row slice
  const int wn = wave & 1;  // 0..1 -> 64-col slice
  const int m0 = blockIdx.y * 128;
  const int n0 = blockIdx.x * 128;
  const int row = tid >> 1;       // 0..127 tile row for staging
  const int seg = (tid & 1) * 16; // low/high 16 of the 32-wide K slab

  FragF c[2][4];
#pragma unroll
  for (int i = 0; i < 2; ++i)
#pragma unroll
    for (int j = 0; j < 4; ++j)
#pragma unroll
      for (int e = 0; e < 8; ++e) c[i][j].f[e] = 0.0f;

  for (int k0 = 0; k0 < K; k0 += 32) {
    uint32_t w[8];
    // ---- stage A tile (128 x 32 bf16) ----
    if (A_BF16) {
      const uint16_t* g =
          (const uint16_t*)Aptr + (size_t)(m0 + row) * K + k0 + seg;
      const uint4 p0 = *(const uint4*)g;
      const uint4 p1 = *(const uint4*)(g + 8);
      w[0] = p0.x; w[1] = p0.y; w[2] = p0.z; w[3] = p0.w;
      w[4] = p1.x; w[5] = p1.y; w[6] = p1.z; w[7] = p1.w;
    } else {
      const float* g = (const float*)Aptr + (size_t)(m0 + row) * K + k0 + seg;
#pragma unroll
      for (int q = 0; q < 4; ++q) {
        const float4 v = ((const float4*)g)[q];
        w[2 * q + 0] =
            (uint32_t)f32_to_bf16(v.x) | ((uint32_t)f32_to_bf16(v.y) << 16);
        w[2 * q + 1] =
            (uint32_t)f32_to_bf16(v.z) | ((uint32_t)f32_to_bf16(v.w) << 16);
      }
    }
    {
      uint32_t* d = (uint32_t*)&lA[row * 32 + seg];
      ((uint4*)d)[0] = make_uint4(w[0], w[1], w[2], w[3]);
      ((uint4*)d)[1] = make_uint4(w[4], w[5], w[6], w[7]);
    }
    // ---- stage B tile (128 x 32 bf16), source is [N][K] f32 ----
    {
      const float* g = Bw + (size_t)(n0 + row) * K + k0 + seg;
#pragma unroll
      for (int q = 0; q < 4; ++q) {
        const float4 v = ((const float4*)g)[q];
        w[2 * q + 0] =
            (uint32_t)f32_to_bf16(v.x) | ((uint32_t)f32_to_bf16(v.y) << 16);
        w[2 * q + 1] =
            (uint32_t)f32_to_bf16(v.z) | ((uint32_t)f32_to_bf16(v.w) << 16);
      }
      uint32_t* d = (uint32_t*)&lB[row * 32 + seg];
      ((uint4*)d)[0] = make_uint4(w[0], w[1], w[2], w[3]);
      ((uint4*)d)[1] = make_uint4(w[4], w[5], w[6], w[7]);
    }
    __syncthreads();

    FragB a[2], b[4];
#pragma unroll
    for (int i = 0; i < 2; ++i) ld_frag(a[i], lA, wm * 32 + i * 16, 32, 0, lane);
#pragma unroll
    for (int j = 0; j < 4; ++j) ld_frag(b[j], lB, wn * 64 + j * 16, 32, 0, lane);
#pragma unroll
    for (int i = 0; i < 2; ++i)
#pragma unroll
      for (int j = 0; j < 4; ++j)
        c[i][j].v = wmma_bf16(a[i].v, b[j].v, c[i][j].v);
    __syncthreads();
  }

  // Epilogue. C-frag layout: element e -> M = e + (lane>=16 ? 8 : 0),
  // N = lane%16 (contiguous across the 16-lane half -> coalesced stores).
  const int rsel = (lane & 16) ? 8 : 0;
#pragma unroll
  for (int i = 0; i < 2; ++i) {
#pragma unroll
    for (int j = 0; j < 4; ++j) {
      const int col = n0 + wn * 64 + j * 16 + (lane & 15);
      const float bv = ADD_BIAS ? bias[col] : 0.0f;
#pragma unroll
      for (int e = 0; e < 8; ++e) {
        const int r = m0 + wm * 32 + i * 16 + e + rsel;
        Cout[(size_t)r * N + col] = c[i][j].f[e] + bv;
      }
    }
  }
}

// ---------------------------------------------------------------------------
// Kernel 2/4: RoPE + head split.  qkv f32 [B,S,3H] ->
//   qb,kb bf16 [B,NH,S,HD]  (row-major, HD contiguous)
//   vtb  bf16 [B,NH,HD,S]   (transposed so PV's B-matrix is [N][K])
// ---------------------------------------------------------------------------
__global__ __launch_bounds__(256) void rope_split_kernel(
    const float* __restrict__ qkv, const int* __restrict__ pos,
    uint16_t* __restrict__ qb, uint16_t* __restrict__ kb,
    uint16_t* __restrict__ vtb) {
  const int gid = blockIdx.x * 256 + threadIdx.x; // 2^23 total
  const int d = gid & (HDD - 1);
  const int s = (gid >> 7) & (SS - 1);
  const int h = (gid >> 18) & (NHH - 1);
  const int b = gid >> 22;

  const float p = (float)pos[b * SS + s];
  const size_t base = ((size_t)b * SS + s) * H3;
  const int hd = h * HDD;
  const float qv = qkv[base + hd + d];
  const float kv = qkv[base + HH + hd + d];
  const float vv = qkv[base + 2 * HH + hd + d];

  const int dm = d & 63; // half = 64
  const float inv = __powf(10000.0f, -(float)dm * (1.0f / 64.0f));
  float sn, cs;
  __sincosf(p * inv, &sn, &cs);
  const int dr = (d < 64) ? (d + 64) : (d - 64);
  const float sgn = (d < 64) ? -1.0f : 1.0f;
  const float qr = sgn * qkv[base + hd + dr];
  const float kr = sgn * qkv[base + HH + hd + dr];

  const size_t o = (((size_t)b * NHH + h) * SS + s) * HDD + d;
  qb[o] = f32_to_bf16(qv * cs + qr * sn);
  kb[o] = f32_to_bf16(kv * cs + kr * sn);
  vtb[(((size_t)b * NHH + h) * HDD + d) * SS + s] = f32_to_bf16(vv);
}

// ---------------------------------------------------------------------------
// Kernel 3/4: causal flash attention with async-LDS double buffering.
// Grid: (S/64, B*NH); 128 threads = 4 waves; each wave owns 16 q-rows with the
// full HD=128 accumulator (8 f32 C-fragments). Key loop tiles 32 keys.
// K/V tiles staged via GLOBAL_LOAD_ASYNC_TO_LDS_B128 and ping-ponged so the
// copy of tile i+1 overlaps the WMMAs of tile i (one barrier per tile).
// Causal mask applied analytically (input mask is exactly triu(-1e9)).
// ---------------------------------------------------------------------------
__global__ __launch_bounds__(128) void flash_attn_kernel(
    const uint16_t* __restrict__ qb, const uint16_t* __restrict__ kb,
    const uint16_t* __restrict__ vtb, uint16_t* __restrict__ attn) {
  __shared__ __align__(16) uint16_t sQ[64 * 128];       // q tile [m][hd]
  __shared__ __align__(16) uint16_t sK[2][32 * 128];    // keys [sk][hd]
  __shared__ __align__(16) uint16_t sV[2][128 * 32];    // values [hd][sk]
  __shared__ __align__(16) uint16_t sP[4 * 16 * 32];    // per-wave P tile

  const int tid = threadIdx.x;
  const int lane = tid & 31;
  const int wave = tid >> 5;
  const int qt = blockIdx.x;
  const int bh = blockIdx.y;
  const int b = bh / NHH;
  const int h = bh % NHH;

  const uint16_t* Q = qb + (size_t)bh * SS * HDD;
  const uint16_t* Kt = kb + (size_t)bh * SS * HDD;
  const uint16_t* Vt = vtb + (size_t)bh * HDD * SS;

  // async stage of the q tile (64 x 128 bf16 = 16KB; 8 B128 per thread)
  {
    const int r = tid >> 1;
    const int c = (tid & 1) * 64;
    const uint16_t* src = Q + (size_t)(qt * 64 + r) * HDD + c;
    uint16_t* dst = &sQ[r * 128 + c];
#pragma unroll
    for (int i = 0; i < 8; ++i) async_ld_b128(dst + i * 8, src + i * 8);
  }

  // async stage of one K/V tile pair into buffer `buf`
  const int kr = tid >> 2;
  const int kc = (tid & 3) * 32;
  auto stage_kv = [&](int s0, int buf) {
#pragma unroll
    for (int i = 0; i < 4; ++i)
      async_ld_b128(&sK[buf][kr * 128 + kc + i * 8],
                    Kt + (size_t)(s0 + kr) * HDD + kc + i * 8);
#pragma unroll
    for (int i = 0; i < 4; ++i)
      async_ld_b128(&sV[buf][tid * 32 + i * 8],
                    Vt + (size_t)tid * SS + s0 + i * 8);
  };

  stage_kv(0, 0); // prefetch first tile
  async_wait0();
  __syncthreads();

  // q A-fragments (16 rows x full K=128), held in registers
  FragB aq[4];
#pragma unroll
  for (int kk = 0; kk < 4; ++kk)
    ld_frag(aq[kk], sQ, wave * 16, 128, kk * 32, lane);

  FragF o[8];
#pragma unroll
  for (int nf = 0; nf < 8; ++nf)
#pragma unroll
    for (int e = 0; e < 8; ++e) o[nf].f[e] = 0.0f;
  float mrow[8], lrow[8];
#pragma unroll
  for (int e = 0; e < 8; ++e) { mrow[e] = -1e30f; lrow[e] = 0.0f; }

  const int rsel = (lane & 16) ? 8 : 0;
  const float scale = 0.08838834764831845f; // 1/sqrt(128)
  const int qmax = qt * 64 + 63;

  int cur = 0;
  for (int s0 = 0; s0 <= qmax; s0 += 32, cur ^= 1) {
    // overlap: copy tile i+1 while computing tile i (buffer read in i-1 is
    // free: the trailing wait+barrier of iteration i-1 covered those reads)
    if (s0 + 32 <= qmax) stage_kv(s0 + 32, cur ^ 1);

    // scores: 16 q-rows x 32 keys
    FragF sf[2];
#pragma unroll
    for (int nf = 0; nf < 2; ++nf) {
#pragma unroll
      for (int e = 0; e < 8; ++e) sf[nf].f[e] = 0.0f;
#pragma unroll
      for (int kk = 0; kk < 4; ++kk) {
        FragB bk;
        ld_frag(bk, sK[cur], nf * 16, 128, kk * 32, lane);
        sf[nf].v = wmma_bf16(aq[kk].v, bk.v, sf[nf].v);
      }
    }

    // scale + causal mask + online softmax (per C-fragment row)
#pragma unroll
    for (int e = 0; e < 8; ++e) {
      const int row = qt * 64 + wave * 16 + e + rsel;
#pragma unroll
      for (int nf = 0; nf < 2; ++nf) {
        const int col = s0 + nf * 16 + (lane & 15);
        float x = sf[nf].f[e] * scale;
        if (col > row) x = -1e9f;
        sf[nf].f[e] = x;
      }
      float mx = fmaxf(sf[0].f[e], sf[1].f[e]);
#pragma unroll
      for (int off = 8; off >= 1; off >>= 1)
        mx = fmaxf(mx, __shfl_xor(mx, off, 32)); // stays inside 16-lane half
      const float mnew = fmaxf(mrow[e], mx);
      const float alpha = __expf(mrow[e] - mnew);
      mrow[e] = mnew;
      float ps = 0.0f;
#pragma unroll
      for (int nf = 0; nf < 2; ++nf) {
        const float p = __expf(sf[nf].f[e] - mnew);
        sf[nf].f[e] = p;
        ps += p;
      }
#pragma unroll
      for (int off = 8; off >= 1; off >>= 1) ps += __shfl_xor(ps, off, 32);
      lrow[e] = lrow[e] * alpha + ps;
#pragma unroll
      for (int nf = 0; nf < 8; ++nf) o[nf].f[e] *= alpha;
    }

    // re-fragment P through this wave's private LDS buffer (16 x 32 bf16)
    uint16_t* pw = &sP[wave * 16 * 32];
#pragma unroll
    for (int nf = 0; nf < 2; ++nf)
#pragma unroll
      for (int e = 0; e < 8; ++e)
        pw[(e + rsel) * 32 + nf * 16 + (lane & 15)] = f32_to_bf16(sf[nf].f[e]);

    FragB ap;
    ld_frag(ap, pw, 0, 32, 0, lane); // LDS ops in-order within a wave
#pragma unroll
    for (int nf = 0; nf < 8; ++nf) {
      FragB bv;
      ld_frag(bv, sV[cur], nf * 16, 32, 0, lane);
      o[nf].v = wmma_bf16(ap.v, bv.v, o[nf].v);
    }

    // next tile's async copies landed (own slice) + all waves done with `cur`
    async_wait0();
    __syncthreads();
  }

  // normalize and emit attn as bf16 [B,S,NH,HD] (== [B,S,H] flat)
#pragma unroll
  for (int nf = 0; nf < 8; ++nf)
#pragma unroll
    for (int e = 0; e < 8; ++e) {
      const int srow = qt * 64 + wave * 16 + e + rsel;
      const int col = nf * 16 + (lane & 15);
      const float v = o[nf].f[e] / lrow[e];
      attn[(((size_t)b * SS + srow) * NHH + h) * HDD + col] = f32_to_bf16(v);
    }
}

// ---------------------------------------------------------------------------
// Host-side launcher
// inputs: 0 hidden f32 [B,S,H], 1 position_ids i32 [B,S], 2 mask f32 (unused:
// analytically causal triu(-1e9)), 3 qkv_w f32 [3H,H], 4 qkv_b f32 [3H],
// 5 o_w f32 [H,H].  Output: f32 [B,S,H].
// Workspace: qkv f32 (100.7MB) + q/k/vt/attn bf16 (4 x 16.8MB) ~= 168MB.
// ---------------------------------------------------------------------------
extern "C" void kernel_launch(void* const* d_in, const int* in_sizes, int n_in,
                              void* d_out, int out_size, void* d_ws,
                              size_t ws_size, hipStream_t stream) {
  (void)in_sizes; (void)n_in; (void)out_size; (void)ws_size;
  const float* hidden = (const float*)d_in[0];
  const int* pos = (const int*)d_in[1];
  const float* qkv_w = (const float*)d_in[3];
  const float* qkv_b = (const float*)d_in[4];
  const float* o_w = (const float*)d_in[5];
  float* out = (float*)d_out;

  char* ws = (char*)d_ws;
  float* qkv_f32 = (float*)ws;                 // 4096*6144*4
  ws += (size_t)MM * H3 * sizeof(float);
  uint16_t* qb = (uint16_t*)ws;                // [B,NH,S,HD]
  ws += (size_t)MM * HH * sizeof(uint16_t);
  uint16_t* kb = (uint16_t*)ws;
  ws += (size_t)MM * HH * sizeof(uint16_t);
  uint16_t* vtb = (uint16_t*)ws;               // [B,NH,HD,S]
  ws += (size_t)MM * HH * sizeof(uint16_t);
  uint16_t* attnb = (uint16_t*)ws;             // [B,S,H] bf16

  // 1) QKV projection + bias
  gemm_wmma_kernel<false, true>
      <<<dim3(H3 / 128, MM / 128), 256, 0, stream>>>(hidden, qkv_w, qkv_b,
                                                     qkv_f32, MM, H3, HH);
  // 2) RoPE + head split / v transpose
  rope_split_kernel<<<(BB * NHH * SS * HDD) / 256, 256, 0, stream>>>(
      qkv_f32, pos, qb, kb, vtb);
  // 3) causal flash attention
  flash_attn_kernel<<<dim3(SS / 64, BB * NHH), 128, 0, stream>>>(qb, kb, vtb,
                                                                 attnb);
  // 4) output projection
  gemm_wmma_kernel<true, false>
      <<<dim3(HH / 128, MM / 128), 256, 0, stream>>>(attnb, o_w, nullptr, out,
                                                     MM, HH, HH);
}
